// BuildCorrelation_86191403696161
// MI455X (gfx1250) — matrix operand order repaired and checked by reading, not compiled, not run on hardware
//
#include <hip/hip_runtime.h>
#include <hip/hip_bf16.h>

// CDNA5 / gfx1250 fused attention + correlation.
// B=64, N=1024, D_IN=64, D=32. Output corr: [64,1024,1024] f32 (256 MB).
// All large matmuls via v_wmma_f32_16x16x32_bf16 (wave32, K=32 == D_QKV).

typedef __attribute__((ext_vector_type(16))) __bf16 v16bf;
typedef __attribute__((ext_vector_type(8)))  __bf16 v8bf;
typedef __attribute__((ext_vector_type(8)))  float  v8f;

#define NB   64
#define NN   1024
#define DIN  64
#define DQ   32

// -------------------------------------------------------------------------
// Kernel 1: QKV projections in f32, emit Q,K bf16 [B,N,32] and V transposed
// bf16 [B,32,N]. 128 threads = 4 rows x 32 out-channels; each thread does
// the Q/K/V dot products (3 x 64 FMA) for one (row, channel).
// -------------------------------------------------------------------------
__global__ __launch_bounds__(128) void qkv_kernel(
    const float* __restrict__ X,
    const float* __restrict__ Wq, const float* __restrict__ bq,
    const float* __restrict__ Wk, const float* __restrict__ bk,
    const float* __restrict__ Wv, const float* __restrict__ bv,
    __bf16* __restrict__ Qbf, __bf16* __restrict__ Kbf,
    __bf16* __restrict__ Vt)
{
    const int t = threadIdx.x;
    const int e = t & 31;                       // output channel 0..31
    const long row = (long)blockIdx.x * 4 + (t >> 5);   // global row in [0, B*N)
    const float* __restrict__ x  = X  + row * DIN;
    const float* __restrict__ wq = Wq + e * DIN;
    const float* __restrict__ wk = Wk + e * DIN;
    const float* __restrict__ wv = Wv + e * DIN;

    float q = bq[e], k = bk[e], v = bv[e];
#pragma unroll
    for (int d = 0; d < DIN; ++d) {
        const float xv = x[d];
        q = fmaf(xv, wq[d], q);
        k = fmaf(xv, wk[d], k);
        v = fmaf(xv, wv[d], v);
    }
    Qbf[row * DQ + e] = (__bf16)q;
    Kbf[row * DQ + e] = (__bf16)k;
    const long b   = row >> 10;
    const long pos = row & (NN - 1);
    Vt[(b * DQ + e) * NN + pos] = (__bf16)v;    // V transposed: [B,32,N]
}

// -------------------------------------------------------------------------
// Kernel 2: one wave (32 threads) per (batch, 16-row strip).
//   S strip [16,1024] via 64 WMMAs -> LDS (f32, 64 KB)
//   f32 softmax (row max / exp / sum, cross-half reduce via shfl_xor)
//   Feature [16,32] via 64 WMMAs (A-op from LDS probs, B-op from Vt)
//   per-row mean/center/inv-norm; store Xc bf16 [B,N,32] and inv f32 [B,N]
// WMMA operand layouts (wave32, 16-bit):
//   A 16x32: lane m=L&15; elems 0..15 = K(16*half + 0..15)  (contiguous, same order as B)
//   B 32x16: lane n=L&15; elems 0..15 = K(16*half + 0..15)  (contiguous)
//   C/D f32 16x16: lane n=L&15; vgpr r = row m = 8*half + r
// -------------------------------------------------------------------------
__global__ __launch_bounds__(32) void attn_kernel(
    const __bf16* __restrict__ Qbf, const __bf16* __restrict__ Kbf,
    const __bf16* __restrict__ Vt,
    __bf16* __restrict__ Xc, float* __restrict__ invn)
{
    __shared__ float sP[16 * NN];               // 64 KB strip

    const int b    = blockIdx.x >> 6;
    const int it   = blockIdx.x & 63;
    const int L    = threadIdx.x;
    const int nl   = L & 15;
    const int half = L >> 4;

    // ---- A-operand: Q tile rows it*16+m ----
    const __bf16* qrow = Qbf + (long)(b * NN + it * 16 + nl) * DQ;
    v16bf aq;
    {
        v8bf lo = *(const v8bf*)(qrow + half * 16);
        v8bf hi = *(const v8bf*)(qrow + half * 16 + 8);
#pragma unroll
        for (int i = 0; i < 8; ++i) { aq[i] = lo[i]; aq[8 + i] = hi[i]; }
    }

    // ---- S = Q K^T / 32 ----
#pragma unroll 4
    for (int jt = 0; jt < 64; ++jt) {
        const __bf16* krow = Kbf + (long)(b * NN + jt * 16 + nl) * DQ + half * 16;
        v16bf bk = *(const v16bf*)krow;
        v8f c = {};
        c = __builtin_amdgcn_wmma_f32_16x16x32_bf16(false, aq, false, bk,
                                                    (short)0, c, false, false);
#pragma unroll
        for (int r = 0; r < 8; ++r)
            sP[(8 * half + r) * NN + jt * 16 + nl] = c[r] * (1.0f / 32.0f);
    }
    asm volatile("s_wait_dscnt 0" ::: "memory");   // cross-lane LDS RAW

    // ---- softmax: lane handles row (L&15), columns [half*512, half*512+512) ----
    {
        float* rowp = sP + (L & 15) * NN + half * 512;
        float mx = -3.4e38f;
        for (int c = 0; c < 512; ++c) mx = fmaxf(mx, rowp[c]);
        mx = fmaxf(mx, __shfl_xor(mx, 16, 32));
        float sum = 0.0f;
        for (int c = 0; c < 512; ++c) {
            float e = __expf(rowp[c] - mx);
            rowp[c] = e;
            sum += e;
        }
        sum += __shfl_xor(sum, 16, 32);
        const float rs = 1.0f / sum;
        for (int c = 0; c < 512; ++c) rowp[c] *= rs;
    }
    asm volatile("s_wait_dscnt 0" ::: "memory");

    // ---- Feature = A @ V  (two 16x16 output tiles over 32 K-steps) ----
    v8f f0 = {}, f1 = {};
#pragma unroll 2
    for (int kt = 0; kt < 32; ++kt) {
        const float* prow = sP + (L & 15) * NN + kt * 32;
        v16bf ap;
#pragma unroll
        for (int i = 0; i < 8; ++i) {
            ap[i]     = (__bf16)prow[half * 16 + i];
            ap[8 + i] = (__bf16)prow[half * 16 + 8 + i];
        }
        const __bf16* v0 = Vt + ((long)b * DQ + nl) * NN + kt * 32 + half * 16;
        v16bf bv0 = *(const v16bf*)v0;
        f0 = __builtin_amdgcn_wmma_f32_16x16x32_bf16(false, ap, false, bv0,
                                                     (short)0, f0, false, false);
        const __bf16* v1 = Vt + ((long)b * DQ + 16 + nl) * NN + kt * 32 + half * 16;
        v16bf bv1 = *(const v16bf*)v1;
        f1 = __builtin_amdgcn_wmma_f32_16x16x32_bf16(false, ap, false, bv1,
                                                     (short)0, f1, false, false);
    }

    // ---- stash Feature tile [16][32] in LDS, compute row stats in f32 ----
#pragma unroll
    for (int r = 0; r < 8; ++r) {
        sP[(8 * half + r) * 32 + nl]      = f0[r];
        sP[(8 * half + r) * 32 + 16 + nl] = f1[r];
    }
    asm volatile("s_wait_dscnt 0" ::: "memory");

    {
        const float* frow = sP + (L & 15) * 32;   // both halves redundantly
        float s = 0.0f;
#pragma unroll
        for (int i = 0; i < 32; ++i) s += frow[i];
        const float mean = s * (1.0f / 32.0f);
        float ss = 0.0f;
        __bf16 xr[32];
#pragma unroll
        for (int i = 0; i < 32; ++i) {
            const float c = frow[i] - mean;
            ss += c * c;
            xr[i] = (__bf16)c;
        }
        const float iv = rsqrtf(ss);
        if (half == 0) {
            const long grow = (long)b * NN + it * 16 + (L & 15);
            __bf16* dst = Xc + grow * DQ;
#pragma unroll
            for (int i = 0; i < 32; ++i) dst[i] = xr[i];
            invn[grow] = iv;
        }
    }
}

// -------------------------------------------------------------------------
// Kernel 3: correlation tiles. One wave per (batch, 16-row strip); the
// A-operand (centered Xc tile) stays in registers across the 64 column
// tiles; each 16x16 tile = ONE bf16 WMMA + f32 scale/clamp + stores.
// This kernel streams the full 256 MB output -> memory-bound by design.
// -------------------------------------------------------------------------
__global__ __launch_bounds__(32) void corr_kernel(
    const __bf16* __restrict__ Xc, const float* __restrict__ invn,
    float* __restrict__ out)
{
    const int b    = blockIdx.x >> 6;
    const int it   = blockIdx.x & 63;
    const int L    = threadIdx.x;
    const int nl   = L & 15;
    const int half = L >> 4;

    const __bf16* arow = Xc + (long)(b * NN + it * 16 + nl) * DQ;
    v16bf a;
    {
        v8bf lo = *(const v8bf*)(arow + half * 16);
        v8bf hi = *(const v8bf*)(arow + half * 16 + 8);
#pragma unroll
        for (int i = 0; i < 8; ++i) { a[i] = lo[i]; a[8 + i] = hi[i]; }
    }
    float invi[8];
#pragma unroll
    for (int r = 0; r < 8; ++r)
        invi[r] = invn[(long)b * NN + it * 16 + 8 * half + r];

#pragma unroll 2
    for (int jt = 0; jt < 64; ++jt) {
        const __bf16* brow = Xc + (long)(b * NN + jt * 16 + nl) * DQ + half * 16;
        if (jt + 1 < 64)
            __builtin_prefetch(brow + DQ * 16, 0, 0);   // next column tile
        v16bf bb = *(const v16bf*)brow;
        const float invj = invn[(long)b * NN + jt * 16 + nl];
        v8f c = {};
        c = __builtin_amdgcn_wmma_f32_16x16x32_bf16(false, a, false, bb,
                                                    (short)0, c, false, false);
        float* orow = out + ((long)b * NN + it * 16 + 8 * half) * NN + jt * 16 + nl;
#pragma unroll
        for (int r = 0; r < 8; ++r) {
            float v = c[r] * invi[r] * invj;
            v = fminf(1.0f, fmaxf(-1.0f, v));
            orow[(long)r * NN] = v;
        }
    }
}

// -------------------------------------------------------------------------
// Host launcher. Workspace layout (bytes):
//   [0,4M)   Qbf  bf16 [B,N,32]
//   [4M,8M)  Kbf  bf16 [B,N,32]
//   [8M,12M) Vt   bf16 [B,32,N]
//   [12M,16M) Xc  bf16 [B,N,32]
//   [16M,16M+256K) invn f32 [B,N]
// -------------------------------------------------------------------------
extern "C" void kernel_launch(void* const* d_in, const int* in_sizes, int n_in,
                              void* d_out, int out_size, void* d_ws, size_t ws_size,
                              hipStream_t stream) {
    const float* X  = (const float*)d_in[0];
    // d_in[1] = EmptyCorrelations (fully overwritten in reference; unused)
    const float* Wq = (const float*)d_in[2];
    const float* bq = (const float*)d_in[3];
    const float* Wk = (const float*)d_in[4];
    const float* bk = (const float*)d_in[5];
    const float* Wv = (const float*)d_in[6];
    const float* bv = (const float*)d_in[7];
    float* out = (float*)d_out;

    char* ws = (char*)d_ws;
    __bf16* Qbf = (__bf16*)(ws);
    __bf16* Kbf = (__bf16*)(ws + (4u  << 20));
    __bf16* Vt  = (__bf16*)(ws + (8u  << 20));
    __bf16* Xc  = (__bf16*)(ws + (12u << 20));
    float*  inv = (float*) (ws + (16u << 20));

    // B*N = 65536 rows; 4 rows per 128-thread block
    qkv_kernel<<<(NB * NN) / 4, 128, 0, stream>>>(X, Wq, bq, Wk, bk, Wv, bv,
                                                  Qbf, Kbf, Vt);
    // one wave per (batch, 16-row strip): 64*64 = 4096 blocks
    attn_kernel<<<NB * (NN / 16), 32, 0, stream>>>(Qbf, Kbf, Vt, Xc, inv);
    corr_kernel<<<NB * (NN / 16), 32, 0, stream>>>(Xc, inv, out);
}
